// Actor_90735479095434
// MI455X (gfx1250) — matrix operand order repaired and checked by reading, compiled, not run on hardware
//
#include <hip/hip_runtime.h>
#include <stdint.h>

// ---------------- problem constants (from reference) ----------------
#define BG       2048          // graphs
#define NNODE    64            // nodes per graph
#define EPG      512           // edges per graph
#define DN       16            // node feature dim
#define DE       64            // node embedding dim
#define DUSER    32
#define DUEMB    64
#define HID      512
#define ROWLEN   (NNODE*DE + DUEMB)   // 4160
#define TOTAL_N  (BG*NNODE)
#define TOTAL_E  (BG*EPG)

typedef __attribute__((ext_vector_type(16))) __bf16 v16bf;
typedef __attribute__((ext_vector_type(8)))  float  v8f;

union FragAB { v16bf bf; unsigned int u[8]; };
union FragC  { v8f f; float x[8]; };

__device__ __forceinline__ unsigned short f2bf(float f) {
    unsigned int u = __float_as_uint(f);
    u += 0x7FFFu + ((u >> 16) & 1u);     // round-to-nearest-even
    return (unsigned short)(u >> 16);
}

// ---------------- kernel 1: per-graph edge aggregation + node MLP ----------------
__global__ void graph_node_kernel(const float* __restrict__ x,
                                  const float* __restrict__ edge_attr,
                                  const int*   __restrict__ edge_index,
                                  const float* __restrict__ W_msg,  const float* __restrict__ b_msg,
                                  const float* __restrict__ W_edge, const float* __restrict__ b_edge,
                                  const float* __restrict__ W_self, const float* __restrict__ b_self,
                                  unsigned short* __restrict__ states)
{
    __shared__ float xl[NNODE * DN];   // graph's node features
    __shared__ float xa[NNODE * DN];   // scatter-summed source features per dst
    __shared__ float ea[NNODE * 2];    // scatter-summed edge attrs per dst
    __shared__ float deg[NNODE];       // in-degree per dst

    const int b = blockIdx.x;
    const int t = threadIdx.x;
    const int nbase = b * NNODE;

    #pragma unroll
    for (int q = 0; q < 4; ++q) {
        int i = t + 256 * q;                       // 1024 = 64*16
        xl[i] = x[(size_t)nbase * DN + i];
        xa[i] = 0.0f;
    }
    if (t < NNODE * 2) ea[t] = 0.0f;
    if (t < NNODE)     deg[t] = 0.0f;
    __syncthreads();

    #pragma unroll
    for (int q = 0; q < 2; ++q) {
        int e = t + 256 * q;                       // 512 edges
        size_t eg = (size_t)b * EPG + e;
        int sl = edge_index[eg] - nbase;                       // src local
        int dl = edge_index[(size_t)TOTAL_E + eg] - nbase;     // dst local
        #pragma unroll
        for (int k = 0; k < DN; ++k)
            atomicAdd(&xa[dl * DN + k], xl[sl * DN + k]);
        atomicAdd(&ea[dl * 2 + 0], edge_attr[eg * 2 + 0]);
        atomicAdd(&ea[dl * 2 + 1], edge_attr[eg * 2 + 1]);
        atomicAdd(&deg[dl], 1.0f);
    }
    __syncthreads();

    #pragma unroll
    for (int q = 0; q < 16; ++q) {
        int o = t + 256 * q;
        int i = o >> 6, j = o & 63;
        float acc = b_self[j] + deg[i] * (b_msg[j] + b_edge[j]);
        #pragma unroll
        for (int k = 0; k < DN; ++k)
            acc += xa[i * DN + k] * W_msg[k * DE + j] + xl[i * DN + k] * W_self[k * DE + j];
        acc += ea[i * 2 + 0] * W_edge[j] + ea[i * 2 + 1] * W_edge[DE + j];
        acc = acc > 0.0f ? acc : 0.0f;
        // deterministic dropout mask (p=0.5, scale 2x)
        unsigned int gid = (unsigned int)((nbase + i) * DE + j);
        unsigned int h = gid * 2654435761u;
        h ^= h >> 16; h *= 2246822519u; h ^= h >> 13;
        float v = (h & 1u) ? acc * 2.0f : 0.0f;
        states[(size_t)b * ROWLEN + o] = f2bf(v);
    }
}

// ---------------- kernel 2: user embedding -> states[:, 4096:4160] ----------------
__global__ void user_kernel(const float* __restrict__ user_s,
                            const float* __restrict__ W1, const float* __restrict__ b1,
                            unsigned short* __restrict__ states)
{
    int o = blockIdx.x * 256 + threadIdx.x;   // 2048*64 outputs
    int bb = o >> 6, j = o & 63;
    float acc = b1[j];
    #pragma unroll
    for (int k = 0; k < DUSER; ++k)
        acc += user_s[(size_t)bb * DUSER + k] * W1[k * DUEMB + j];
    acc = acc > 0.0f ? acc : 0.0f;
    states[(size_t)bb * ROWLEN + NNODE * DE + j] = f2bf(acc);
}

// ---------------- kernel 3: f32 -> bf16 weight conversion ----------------
__global__ void cvt_bf16(const float* __restrict__ s, unsigned short* __restrict__ d, int n)
{
    int i = blockIdx.x * 256 + threadIdx.x;
    if (i < n) d[i] = f2bf(s[i]);
}

// ---------------- kernel 4: bf16 WMMA GEMM, C = relu?(A@B + bias) ----------------
// A: [M,K] bf16 row-major, B: [K,N] bf16 row-major.
// Block tile 64x64, BK=64, 8 waves (4M x 2N), each wave 16x32 (2 accumulators).
// Register-staged LDS double buffering: 1 barrier / K-step, 4 WMMA / wave / K-step.
__global__ void gemm_bf16_wmma(const unsigned short* __restrict__ A,
                               const unsigned short* __restrict__ Bm,
                               const float* __restrict__ bias,
                               unsigned short* __restrict__ Obf,
                               float* __restrict__ Of,
                               int M, int K, int N, int do_relu)
{
    __shared__ unsigned short As[2][64 * 64];   // 8KB per buffer
    __shared__ unsigned short Bs[2][64 * 64];   // 8KB per buffer

    const int t    = threadIdx.x;
    const int lane = t & 31;
    const int wave = t >> 5;                 // 8 waves
    const int m0   = blockIdx.x * 64;
    const int n0   = blockIdx.y * 64;
    const int wm   = (wave & 3) * 16;        // wave M offset in tile: 0..48
    const int wn   = (wave >> 2) * 32;       // wave N offset in tile: 0,32
    const int mrow = lane & 15;
    const int h    = lane >> 4;

    // ---- staging pointers (hoisted; constant stride bumps in the loop) ----
    // tile = 64 rows x 64 bf16 = 512 uint4 slots; 256 threads x 2 slots.
    const int slot0 = t, slot1 = t + 256;
    const int rA0 = slot0 >> 3, cA0 = slot0 & 7;
    const int rA1 = slot1 >> 3, cA1 = slot1 & 7;
    const uint4* aSrc0 = (const uint4*)(A + (size_t)(m0 + rA0) * K) + cA0;
    const uint4* aSrc1 = (const uint4*)(A + (size_t)(m0 + rA1) * K) + cA1;
    const uint4* bSrc0 = (const uint4*)(Bm + (size_t)rA0 * N + n0) + cA0;
    const uint4* bSrc1 = (const uint4*)(Bm + (size_t)rA1 * N + n0) + cA1;
    const size_t bStride = (size_t)8 * N;    // 64 rows * N bf16, in uint4 units

    FragC acc0, acc1;
    #pragma unroll
    for (int r = 0; r < 8; ++r) { acc0.x[r] = 0.0f; acc1.x[r] = 0.0f; }

    // prologue: stage tile 0 into buffer 0
    {
        uint4 a0 = *aSrc0, a1 = *aSrc1, b0 = *bSrc0, b1 = *bSrc1;
        ((uint4*)As[0])[slot0] = a0; ((uint4*)As[0])[slot1] = a1;
        ((uint4*)Bs[0])[slot0] = b0; ((uint4*)Bs[0])[slot1] = b1;
    }
    __syncthreads();

    int p = 0;
    for (int kk = 0; kk < K; kk += 64) {
        const bool last = (kk + 64 >= K);
        uint4 na0, na1, nb0, nb1;
        if (!last) {
            aSrc0 += 8; aSrc1 += 8;          // +64 bf16 along K
            bSrc0 += bStride; bSrc1 += bStride;
            na0 = *aSrc0; na1 = *aSrc1; nb0 = *bSrc0; nb1 = *bSrc1;
            __builtin_prefetch((const void*)(aSrc0 + 8), 0, 1);
            __builtin_prefetch((const void*)(bSrc0 + bStride), 0, 1);
        }

        // compute: 2 K-substeps x 2 N-accumulators = 4 WMMA (A frag reused)
        const unsigned short* __restrict__ Ab = As[p];
        const unsigned short* __restrict__ Bb = Bs[p];
        #pragma unroll
        for (int s = 0; s < 2; ++s) {
            FragAB a, b0f, b1f;
            const int m = wm + mrow;
            #pragma unroll
            for (int v = 0; v < 8; ++v) {
                int k0 = (v < 4) ? (8 * h + 2 * v) : (16 + 8 * h + 2 * (v - 4));
                a.u[v] = *(const unsigned int*)&Ab[m * 64 + 32 * s + k0];
            }
            const int bcol = wn + mrow;
            #pragma unroll
            for (int v = 0; v < 8; ++v) {
                int k0 = (v < 4) ? (8 * h + 2 * v) : (16 + 8 * h + 2 * (v - 4));
                int r0 = (32 * s + k0) * 64, r1 = r0 + 64;
                b0f.u[v] = (unsigned int)Bb[r0 + bcol] | ((unsigned int)Bb[r1 + bcol] << 16);
                b1f.u[v] = (unsigned int)Bb[r0 + bcol + 16] | ((unsigned int)Bb[r1 + bcol + 16] << 16);
            }
            acc0.f = __builtin_amdgcn_wmma_f32_16x16x32_bf16(
                false, a.bf, false, b0f.bf, (short)0, acc0.f, false, false);
            acc1.f = __builtin_amdgcn_wmma_f32_16x16x32_bf16(
                false, a.bf, false, b1f.bf, (short)0, acc1.f, false, false);
        }

        if (!last) {
            ((uint4*)As[p ^ 1])[slot0] = na0; ((uint4*)As[p ^ 1])[slot1] = na1;
            ((uint4*)Bs[p ^ 1])[slot0] = nb0; ((uint4*)Bs[p ^ 1])[slot1] = nb1;
        }
        __syncthreads();
        p ^= 1;
    }

    // epilogue: D layout — VGPR r: M = r + 8h, N = lane%16
    const int n_a = n0 + wn + mrow;
    const int n_b = n_a + 16;
    const float bsa = bias ? bias[n_a] : 0.0f;
    const float bsb = bias ? bias[n_b] : 0.0f;
    #pragma unroll
    for (int r = 0; r < 8; ++r) {
        int m = m0 + wm + r + 8 * h;
        float va = acc0.x[r] + bsa;
        float vb = acc1.x[r] + bsb;
        if (do_relu) { va = va > 0.0f ? va : 0.0f; vb = vb > 0.0f ? vb : 0.0f; }
        if (Obf) {
            Obf[(size_t)m * N + n_a] = f2bf(va);
            Obf[(size_t)m * N + n_b] = f2bf(vb);
        } else {
            Of[(size_t)m * N + n_a] = va;
            Of[(size_t)m * N + n_b] = vb;
        }
    }
}

// ---------------- kernel 5: softmax over the BATCH axis (axis 0 of [B,N]) ----------------
__global__ void softmax_col(const float* __restrict__ logits, float* __restrict__ out)
{
    __shared__ float red[256];
    const int n = blockIdx.x;      // 64 columns
    const int t = threadIdx.x;

    float mx = -1e30f;
    for (int b = t; b < BG; b += 256) mx = fmaxf(mx, logits[(size_t)b * NNODE + n]);
    red[t] = mx; __syncthreads();
    for (int s = 128; s > 0; s >>= 1) { if (t < s) red[t] = fmaxf(red[t], red[t + s]); __syncthreads(); }
    mx = red[0]; __syncthreads();

    float sm = 0.0f;
    for (int b = t; b < BG; b += 256) sm += __expf(logits[(size_t)b * NNODE + n] - mx);
    red[t] = sm; __syncthreads();
    for (int s = 128; s > 0; s >>= 1) { if (t < s) red[t] += red[t + s]; __syncthreads(); }
    float inv = 1.0f / red[0];

    for (int b = t; b < BG; b += 256)
        out[(size_t)b * NNODE + n] = __expf(logits[(size_t)b * NNODE + n] - mx) * inv;
}

// ---------------- launcher ----------------
extern "C" void kernel_launch(void* const* d_in, const int* in_sizes, int n_in,
                              void* d_out, int out_size, void* d_ws, size_t ws_size,
                              hipStream_t stream)
{
    const float* x         = (const float*)d_in[0];
    const float* edge_attr = (const float*)d_in[1];
    const float* user_s    = (const float*)d_in[2];
    const float* W_msg     = (const float*)d_in[3];
    const float* b_msg     = (const float*)d_in[4];
    const float* W_edge    = (const float*)d_in[5];
    const float* b_edge    = (const float*)d_in[6];
    const float* W_self    = (const float*)d_in[7];
    const float* b_self    = (const float*)d_in[8];
    const float* W1        = (const float*)d_in[9];
    const float* b1        = (const float*)d_in[10];
    const float* W2        = (const float*)d_in[11];
    const float* b2        = (const float*)d_in[12];
    const float* W3        = (const float*)d_in[13];
    const float* b3        = (const float*)d_in[14];
    const float* W4        = (const float*)d_in[15];
    const float* b4        = (const float*)d_in[16];
    const int*   edge_index= (const int*)d_in[17];

    // workspace carve-out (256B aligned)
    char* ws = (char*)d_ws;
    auto take = [&](size_t bytes) {
        char* p = ws;
        ws += (bytes + 255) & ~(size_t)255;
        return p;
    };
    unsigned short* states = (unsigned short*)take((size_t)BG * ROWLEN * 2);     // 17 MB
    unsigned short* W2b    = (unsigned short*)take((size_t)ROWLEN * HID * 2);    // 4.3 MB
    unsigned short* W3b    = (unsigned short*)take((size_t)HID * (HID/2) * 2);
    unsigned short* W4b    = (unsigned short*)take((size_t)(HID/2) * NNODE * 2);
    unsigned short* h1     = (unsigned short*)take((size_t)BG * HID * 2);
    unsigned short* h2     = (unsigned short*)take((size_t)BG * (HID/2) * 2);
    float*          logits = (float*)take((size_t)BG * NNODE * 4);

    // phase 1: build states (bf16) — graph conv + user MLP, weight conversion
    graph_node_kernel<<<BG, 256, 0, stream>>>(x, edge_attr, edge_index,
                                              W_msg, b_msg, W_edge, b_edge,
                                              W_self, b_self, states);
    user_kernel<<<(BG * DUEMB) / 256, 256, 0, stream>>>(user_s, W1, b1, states);
    cvt_bf16<<<(ROWLEN * HID + 255) / 256, 256, 0, stream>>>(W2, W2b, ROWLEN * HID);
    cvt_bf16<<<(HID * (HID/2) + 255) / 256, 256, 0, stream>>>(W3, W3b, HID * (HID/2));
    cvt_bf16<<<((HID/2) * NNODE + 255) / 256, 256, 0, stream>>>(W4, W4b, (HID/2) * NNODE);

    // phase 2: MLP chain via bf16 WMMA (block tile 64x64)
    gemm_bf16_wmma<<<dim3(BG / 64, HID / 64), 256, 0, stream>>>(
        states, W2b, b2, h1, nullptr, BG, ROWLEN, HID, 1);
    gemm_bf16_wmma<<<dim3(BG / 64, (HID/2) / 64), 256, 0, stream>>>(
        h1, W3b, b3, h2, nullptr, BG, HID, HID / 2, 1);
    gemm_bf16_wmma<<<dim3(BG / 64, NNODE / 64), 256, 0, stream>>>(
        h2, W4b, b4, nullptr, logits, BG, HID / 2, NNODE, 0);

    // phase 3: softmax across the batch axis, straight into d_out [1, B, N]
    softmax_col<<<NNODE, 256, 0, stream>>>(logits, (float*)d_out);
}
